// SeqAttention_16363825398642
// MI455X (gfx1250) — matrix operand order repaired and checked by reading, not compiled, not run on hardware
//
#include <hip/hip_runtime.h>

// ---------------------------------------------------------------------------
// Adaptive-span attention, fused, CDNA5 (gfx1250) wave32 + WMMA f16->f32.
//   B=128 (8 seq x 16 heads), M=512, L=1024, H=64, RAMP=32
// out[b,i,:] = sum_j w[b,i,j] * V[b,i+j,:]
//   w = e^{s-m} * mask / (sum e*mask + 1e-8 * sum e),  s = (Q.K + Q.PE)/8
// ---------------------------------------------------------------------------

typedef __attribute__((ext_vector_type(16))) _Float16 v16h;
typedef __attribute__((ext_vector_type(8)))  float    v8f;

#define NB_HEADS 16
#define HEAD_DIM 64
#define ATTN_LIM 1024
#define BLOCK    512
#define BATCH    8
#define BH       (BATCH * NB_HEADS)     // 128
#define KVLEN    (BLOCK + ATTN_LIM)     // 1536
#define ROWS     16                     // query rows per workgroup
#define WPAD     1056                   // padded absolute span (33 chunks of 32)

union Frag16 { v16h v; uint4 q[2]; };

// A-fragment (16x32 f16): lane<16 -> M=lane, K in {kb..kb+7, kb+16..kb+23};
// lane>=16 -> same M, K ranges shifted by +8. Two 16B contiguous loads.
__device__ __forceinline__ v16h load_a(const _Float16* rowbase, int kbase, int hi) {
    Frag16 f;
    const _Float16* p = rowbase + kbase + 8 * hi;
    f.q[0] = *reinterpret_cast<const uint4*>(p);
    f.q[1] = *reinterpret_cast<const uint4*>(p + 16);
    return f.v;
}

// B-fragment (32x16 f16): lane = column N; lanes<16 hold K in {0..7, 16..23},
// lanes>=16 hold K in {8..15, 24..31} (matches load_a). Two 16B loads.
__device__ __forceinline__ v16h load_b(const _Float16* colbase, int kbase, int hi) {
    Frag16 f;
    const _Float16* p = colbase + kbase + 8 * hi;
    f.q[0] = *reinterpret_cast<const uint4*>(p);
    f.q[1] = *reinterpret_cast<const uint4*>(p + 16);
    return f.v;
}

#define WMMA_F16(A, B, C) \
    __builtin_amdgcn_wmma_f32_16x16x32_f16(false, (A), false, (B), (short)0, (C), false, false)

// ---------------------------------------------------------------------------
// Pre-pass: fp32 -> fp16 elementwise convert (Q, K keep row-major [.,.,H]).
// ---------------------------------------------------------------------------
__global__ void cvt_f16_kernel(const float* __restrict__ in, _Float16* __restrict__ out, size_t n) {
    size_t i = (size_t)blockIdx.x * blockDim.x + threadIdx.x;
    size_t stride = (size_t)gridDim.x * blockDim.x;
    for (; i < n; i += stride) out[i] = (_Float16)in[i];
}

// ---------------------------------------------------------------------------
// Pre-pass: tiled transpose + convert: in [R][C] f32 -> out [C][R] f16.
// blockIdx.z = batch slab of R*C elements. Used for V (R=1536,C=64) and
// PE (R=64,C=1024). 32x32 tiles, 256 threads.
// ---------------------------------------------------------------------------
__global__ void transpose_cvt_kernel(const float* __restrict__ in, _Float16* __restrict__ out,
                                     int R, int C) {
    __shared__ float tile[32][33];
    const size_t off = (size_t)blockIdx.z * (size_t)R * (size_t)C;
    const int tx = threadIdx.x & 31;
    const int ty = threadIdx.x >> 5;   // 0..7
    const int r0 = blockIdx.y * 32;
    const int c0 = blockIdx.x * 32;
#pragma unroll
    for (int k = 0; k < 4; ++k) {
        int r = ty + 8 * k;
        tile[r][tx] = in[off + (size_t)(r0 + r) * C + (c0 + tx)];
    }
    __syncthreads();
#pragma unroll
    for (int k = 0; k < 4; ++k) {
        int c = ty + 8 * k;
        out[off + (size_t)(c0 + c) * R + (r0 + tx)] = (_Float16)tile[tx][c];
    }
}

// ---------------------------------------------------------------------------
// Main fused kernel. grid = (M/16, B), block = 256 (8 wave32).
//   Qh : [B][M][H]      f16
//   Kh : [B][M+L][H]    f16
//   Vt : [B][H][M+L]    f16 (transposed)
//   PEt: [L][H]         f16 (transposed)
// ---------------------------------------------------------------------------
__launch_bounds__(256)
__global__ void attn_main_kernel(const _Float16* __restrict__ Qh,
                                 const _Float16* __restrict__ Kh,
                                 const _Float16* __restrict__ Vt,
                                 const _Float16* __restrict__ PEt,
                                 const float*    __restrict__ span_val,
                                 float*          __restrict__ out) {
    // 65536 B scores (f32 [16][1024]) + 33792 B skewed weights (f16 [16][1056])
    __shared__ __align__(16) unsigned char smem[ROWS * ATTN_LIM * 4 + ROWS * WPAD * 2];
    float*    S     = (float*)smem;                            // [16][1024]
    _Float16* W     = (_Float16*)(smem + ROWS * ATTN_LIM * 4); // [16][1056]
    float*    Opart = (float*)smem;                            // overlay: [4][16][16]

    const int b    = blockIdx.y;
    const int i0   = blockIdx.x * ROWS;
    const int tid  = threadIdx.x;
    const int wave = tid >> 5;
    const int lane = tid & 31;
    const int n16  = lane & 15;
    const int hi   = lane >> 4;

    // zero the skewed-weight tile (read later with padded columns)
    {
        unsigned int* wz = (unsigned int*)W;
        for (int i = tid; i < ROWS * WPAD / 2; i += 256) wz[i] = 0u;
    }

    // Q tile A-fragments (shared by both score GEMMs): rows i0..i0+15, K = H
    const _Float16* qbase = Qh + ((size_t)b * BLOCK + i0 + n16) * HEAD_DIM;
    const v16h qa0 = load_a(qbase, 0, hi);
    const v16h qa1 = load_a(qbase, 32, hi);

    // ---- GEMM2: positional scores in relative coords -> plain store to S ----
    for (int cc = wave; cc < ATTN_LIM / 16; cc += 8) {
        const int j0 = cc * 16;
        const _Float16* brow = PEt + (size_t)(j0 + n16) * HEAD_DIM;
        v8f acc = {};
        acc = WMMA_F16(qa0, load_b(brow, 0, hi), acc);
        acc = WMMA_F16(qa1, load_b(brow, 32, hi), acc);
#pragma unroll
        for (int v = 0; v < 8; ++v) {
            const int r = v + 8 * hi;
            S[r * ATTN_LIM + j0 + n16] = acc[v];
        }
    }
    __syncthreads();

    // ---- GEMM1: content scores in absolute coords -> skewed add into S ----
    // keys t = i0 + 16c + n16; 65 chunks cover t-i0 in [0,1040) >= 15+1023.
    const _Float16* kb = Kh + (size_t)b * KVLEN * HEAD_DIM;
    for (int c = wave; c < 65; c += 8) {
        const int t = i0 + c * 16 + n16;        // always < 1536
        const _Float16* brow = kb + (size_t)t * HEAD_DIM;
        v8f acc = {};
        acc = WMMA_F16(qa0, load_b(brow, 0, hi), acc);
        acc = WMMA_F16(qa1, load_b(brow, 32, hi), acc);
#pragma unroll
        for (int v = 0; v < 8; ++v) {
            const int r = v + 8 * hi;
            const int j = c * 16 + n16 - r;     // relative coordinate
            if (j >= 0 && j < ATTN_LIM) S[r * ATTN_LIM + j] += acc[v];
        }
    }
    __syncthreads();

    // ---- softmax + adaptive-span mask (folded) + renorm; write skewed f16 ----
    {
        float z = span_val[b & (NB_HEADS - 1)];
        z = fminf(fmaxf(z, 0.0f), 1.0f);
        const float zoff = z * (float)ATTN_LIM - (float)(ATTN_LIM - 1);
        for (int rr = 0; rr < 2; ++rr) {
            const int r = wave * 2 + rr;
            float* srow = S + r * ATTN_LIM;
            float m = -3.4e38f;
            for (int jj = lane; jj < ATTN_LIM; jj += 32) m = fmaxf(m, srow[jj]);
            for (int d = 16; d >= 1; d >>= 1) m = fmaxf(m, __shfl_xor(m, d, 32));
            m *= 0.125f;                        // 1/sqrt(H)
            float sz = 0.0f, szm = 0.0f;
            for (int jj = lane; jj < ATTN_LIM; jj += 32) {
                const float e = __expf(srow[jj] * 0.125f - m);
                srow[jj] = e;
                const float msk = fminf(fmaxf(fmaf((float)jj + zoff, 1.0f / 32.0f, 1.0f), 0.0f), 1.0f);
                sz += e;
                szm += e * msk;
            }
            for (int d = 16; d >= 1; d >>= 1) {
                sz  += __shfl_xor(sz,  d, 32);
                szm += __shfl_xor(szm, d, 32);
            }
            const float inv = 1.0f / (szm + 1e-8f * sz);
            _Float16* wrow = W + r * WPAD + r;  // skew: absolute column = r + jj
            for (int jj = lane; jj < ATTN_LIM; jj += 32) {
                const float msk = fminf(fmaxf(fmaf((float)jj + zoff, 1.0f / 32.0f, 1.0f), 0.0f), 1.0f);
                wrow[jj] = (_Float16)(srow[jj] * msk * inv);
            }
        }
    }
    __syncthreads();

    // ---- GEMM3: out(16x64) = W_abs(16x1056) @ V(1056x64), absolute coords ----
    // 4 N-tiles of 16 (H) x 2 K-halves -> 8 waves; 33 K-chunks of 32.
    const int ntile = wave >> 1;
    const int khalf = wave & 1;
    const int kc0 = khalf ? 17 : 0;
    const int kc1 = khalf ? 33 : 17;
    const _Float16* wrow = W + (size_t)n16 * WPAD;                                  // A rows
    const _Float16* vrow = Vt + ((size_t)b * HEAD_DIM + ntile * 16 + n16) * KVLEN + i0; // B cols
    v8f oacc = {};
    for (int kc = kc0; kc < kc1; ++kc) {
        const v16h a  = load_a(wrow, kc * 32, hi);
        const v16h bb = load_b(vrow, kc * 32, hi);
        oacc = WMMA_F16(a, bb, oacc);
    }
    if (khalf == 1) {
#pragma unroll
        for (int v = 0; v < 8; ++v)
            Opart[ntile * 256 + (v + 8 * hi) * 16 + n16] = oacc[v];
    }
    __syncthreads();
    if (khalf == 0) {
#pragma unroll
        for (int v = 0; v < 8; ++v) {
            const int r = v + 8 * hi;
            const float val = oacc[v] + Opart[ntile * 256 + r * 16 + n16];
            out[((size_t)b * BLOCK + i0 + r) * HEAD_DIM + ntile * 16 + n16] = val;
        }
    }
}

// ---------------------------------------------------------------------------
extern "C" void kernel_launch(void* const* d_in, const int* in_sizes, int n_in,
                              void* d_out, int out_size, void* d_ws, size_t ws_size,
                              hipStream_t stream) {
    const float* Q  = (const float*)d_in[0];  // [128][512][64]
    const float* K  = (const float*)d_in[1];  // [128][1536][64]
    const float* V  = (const float*)d_in[2];  // [128][1536][64]
    const float* PE = (const float*)d_in[3];  // [1][64][1024]
    const float* SV = (const float*)d_in[4];  // [16]
    float* out = (float*)d_out;               // [128][512][64] f32

    const size_t qn = (size_t)BH * BLOCK * HEAD_DIM;   //  4,194,304
    const size_t kn = (size_t)BH * KVLEN * HEAD_DIM;   // 12,582,912
    const size_t pn = (size_t)HEAD_DIM * ATTN_LIM;     //     65,536

    char* ws = (char*)d_ws;
    _Float16* Qh  = (_Float16*)(ws);
    _Float16* Kh  = (_Float16*)(ws + qn * 2);
    _Float16* Vt  = (_Float16*)(ws + qn * 2 + kn * 2);
    _Float16* PEt = (_Float16*)(ws + qn * 2 + kn * 2 + kn * 2);
    // total ws use: 2*(qn + 2*kn + pn) bytes ~= 56.2 MB

    cvt_f16_kernel<<<2048, 256, 0, stream>>>(Q, Qh, qn);
    cvt_f16_kernel<<<4096, 256, 0, stream>>>(K, Kh, kn);
    transpose_cvt_kernel<<<dim3(HEAD_DIM / 32, KVLEN / 32, BH), 256, 0, stream>>>(V, Vt, KVLEN, HEAD_DIM);
    transpose_cvt_kernel<<<dim3(ATTN_LIM / 32, HEAD_DIM / 32, 1), 256, 0, stream>>>(PE, PEt, HEAD_DIM, ATTN_LIM);
    attn_main_kernel<<<dim3(BLOCK / ROWS, BH), 256, 0, stream>>>(Qh, Kh, Vt, PEt, SV, out);
}
